// LowBitEncoder_43576738185534
// MI455X (gfx1250) — matrix operand order repaired and checked by reading, not compiled, not run on hardware
//
#include <hip/hip_runtime.h>
#include <cmath>

typedef __attribute__((ext_vector_type(16))) __bf16 v16bf;
typedef __attribute__((ext_vector_type(8)))  float  v8f;

#define BDIM  8
#define SDIM  2048
#define DIN   4096
#define DOUT  4096
#define MROWS 32            // rows per block (2 x 16-row WMMA M-tiles)
#define TPB   512           // 16 waves (wave32) -> 4 waves per SIMD32
#define THRESH 0.1f
#define LN_EPS 1e-5f

// ---------------------------------------------------------------------------
// Fragment container: 32 bytes = 16 bf16 = two b128 global loads
// ---------------------------------------------------------------------------
union ABFrag {
    uint4 u[2];
    v16bf v;
};

__device__ __forceinline__ unsigned short f2bf_rne(float f) {
    unsigned u = __float_as_uint(f);
    unsigned r = u + 0x7FFFu + ((u >> 16) & 1u);   // round-to-nearest-even
    return (unsigned short)(r >> 16);
}

// ---------------------------------------------------------------------------
// x (fp32) -> bf16, 8 elements per thread, b128 in / b128 out
// ---------------------------------------------------------------------------
__global__ void cvt_x_bf16_kernel(const float* __restrict__ x,
                                  unsigned short* __restrict__ xb,
                                  int n8) {
    int i = blockIdx.x * blockDim.x + threadIdx.x;
    if (i >= n8) return;
    const float4* xv = (const float4*)x;
    float4 a = xv[2 * i + 0];
    float4 b = xv[2 * i + 1];
    uint4 o;
    o.x = (unsigned)f2bf_rne(a.x) | ((unsigned)f2bf_rne(a.y) << 16);
    o.y = (unsigned)f2bf_rne(a.z) | ((unsigned)f2bf_rne(a.w) << 16);
    o.z = (unsigned)f2bf_rne(b.x) | ((unsigned)f2bf_rne(b.y) << 16);
    o.w = (unsigned)f2bf_rne(b.z) | ((unsigned)f2bf_rne(b.w) << 16);
    ((uint4*)xb)[i] = o;
}

// ---------------------------------------------------------------------------
// W (fp32) -> ternary {-1,0,+1} in bf16 (exact), same [DOUT, DIN] layout
// ---------------------------------------------------------------------------
__device__ __forceinline__ unsigned short tern_bf16(float w) {
    if (fabsf(w) < THRESH) return 0;
    return (w > 0.0f) ? (unsigned short)0x3F80 : (unsigned short)0xBF80;
}

__global__ void ternarize_w_kernel(const float* __restrict__ w,
                                   unsigned short* __restrict__ wb,
                                   int n8) {
    int i = blockIdx.x * blockDim.x + threadIdx.x;
    if (i >= n8) return;
    const float4* wv = (const float4*)w;
    float4 a = wv[2 * i + 0];
    float4 b = wv[2 * i + 1];
    uint4 o;
    o.x = (unsigned)tern_bf16(a.x) | ((unsigned)tern_bf16(a.y) << 16);
    o.y = (unsigned)tern_bf16(a.z) | ((unsigned)tern_bf16(a.w) << 16);
    o.z = (unsigned)tern_bf16(b.x) | ((unsigned)tern_bf16(b.y) << 16);
    o.w = (unsigned)tern_bf16(b.z) | ((unsigned)tern_bf16(b.w) << 16);
    ((uint4*)wb)[i] = o;
}

// ---------------------------------------------------------------------------
// Fused ternary-GEMM (bf16 WMMA) + bias/scale + LayerNorm + tanh-quantize.
// One block = 32 output rows x full DOUT. 512 threads = 16 waves (wave32),
// 4 waves/SIMD32 for latency hiding at the 1-block/WGP LDS occupancy.
// Wave w owns columns [w*256, w*256+256), as 4 chunks of 64 (2 M-tiles x
// 4 N-tiles of 16x16 accumulators; each B fragment feeds 2 WMMAs).
// y tile staged in LDS as bf16: 32 x 4096 x 2B = 256 KB (< 320 KB/WGP).
// LayerNorm stats computed f32 from the staged tile; quantize in-block.
// ---------------------------------------------------------------------------
__global__ __launch_bounds__(TPB, 1)
void bitnet_gemm_ln_kernel(const unsigned short* __restrict__ xb_u,
                           const unsigned short* __restrict__ wb_u,
                           const float* __restrict__ bias,
                           const float* __restrict__ scale,
                           const float* __restrict__ gamma,
                           const float* __restrict__ beta,
                           const float* __restrict__ qscale,
                           float* __restrict__ out) {
    extern __shared__ unsigned short ytile[];      // MROWS * DOUT bf16
    const __bf16* __restrict__ xb = (const __bf16*)xb_u;
    const __bf16* __restrict__ wb = (const __bf16*)wb_u;

    const int tid  = threadIdx.x;
    const int lane = tid & 31;
    const int wave = tid >> 5;                     // 0..15
    const int nlo  = lane & 15;                    // N / M index within half-wave
    const int hi   = lane >> 4;                    // half-wave selector
    const int m0   = blockIdx.x * MROWS;

    // A-fragment bases: lane nlo holds row (m0 + mt*16 + nlo); low half-wave
    // K{0..7,16..23}, high half K{8..15,24..31}   (ISA 7.12.2, 16-bit A 16x32)
    const __bf16* parow0 = xb + (long long)(m0 + nlo) * DIN;
    const __bf16* parow1 = xb + (long long)(m0 + 16 + nlo) * DIN;

    const v8f vzero = {0.f, 0.f, 0.f, 0.f, 0.f, 0.f, 0.f, 0.f};

    for (int chunk = 0; chunk < 4; ++chunk) {
        const int nbase = wave * 256 + chunk * 64;
        v8f acc[2][4];
#pragma unroll
        for (int mt = 0; mt < 2; ++mt)
#pragma unroll
            for (int t = 0; t < 4; ++t) acc[mt][t] = vzero;

        for (int k0 = 0; k0 < DIN; k0 += 32) {
            ABFrag a0, a1;
            a0.u[0] = *(const uint4*)(parow0 + k0 + hi * 8);
            a0.u[1] = *(const uint4*)(parow0 + k0 + 16 + hi * 8);
            a1.u[0] = *(const uint4*)(parow1 + k0 + hi * 8);
            a1.u[1] = *(const uint4*)(parow1 + k0 + 16 + hi * 8);

#pragma unroll
            for (int t = 0; t < 4; ++t) {
                // B-fragment: lane nlo = column N; low half K{0..7,16..23}, high half
                // K{8..15,24..31} along DIN in row-major [DOUT,DIN] W
                const __bf16* pb =
                    wb + (long long)(nbase + t * 16 + nlo) * DIN + k0 + hi * 8;
                ABFrag b;
                b.u[0] = *(const uint4*)(pb);
                b.u[1] = *(const uint4*)(pb + 16);

                acc[0][t] = __builtin_amdgcn_wmma_f32_16x16x32_bf16(
                    false, a0.v, false, b.v, (short)0, acc[0][t], false, false);
                acc[1][t] = __builtin_amdgcn_wmma_f32_16x16x32_bf16(
                    false, a1.v, false, b.v, (short)0, acc[1][t], false, false);
            }
        }

        // Epilogue: y = (acc + bias) * scale  -> bf16 LDS tile
        // C/D layout: VGPR r, lane: M = r + 8*hi, N = nlo  (ISA 7.12.2)
#pragma unroll
        for (int t = 0; t < 4; ++t) {
            const int n  = nbase + t * 16 + nlo;
            const float sc = scale[n];
            const float bb = bias[n] * sc;
#pragma unroll
            for (int mt = 0; mt < 2; ++mt) {
#pragma unroll
                for (int r = 0; r < 8; ++r) {
                    const int m = mt * 16 + hi * 8 + r;
                    ytile[m * DOUT + n] = f2bf_rne(acc[mt][t][r] * sc + bb);
                }
            }
        }
    }

    __syncthreads();

    // ---- LayerNorm over DOUT + tanh quantization -------------------------
    // 16-lane group g owns row g (32 rows x 16 lanes = 512 threads).
    // Each lane handles 4 consecutive columns per step: b64 LDS reads,
    // b128 gamma/beta loads and b128 output stores. wave32 shuffle width 16.
    const int g  = tid >> 4;                       // row 0..31
    const int lt = tid & 15;                       // lane within group
    const unsigned short* row = ytile + g * DOUT;

    float s = 0.f, s2 = 0.f;
    for (int j = lt * 4; j < DOUT; j += 64) {
        uint2 pk = *(const uint2*)(row + j);       // 4 bf16
        float v0 = __uint_as_float(pk.x << 16);
        float v1 = __uint_as_float(pk.x & 0xFFFF0000u);
        float v2 = __uint_as_float(pk.y << 16);
        float v3 = __uint_as_float(pk.y & 0xFFFF0000u);
        s  += (v0 + v1) + (v2 + v3);
        s2 += (v0 * v0 + v1 * v1) + (v2 * v2 + v3 * v3);
    }
#pragma unroll
    for (int off = 8; off >= 1; off >>= 1) {
        s  += __shfl_xor(s,  off, 16);
        s2 += __shfl_xor(s2, off, 16);
    }
    const float mean = s * (1.0f / DOUT);
    const float var  = s2 * (1.0f / DOUT) - mean * mean;
    const float rstd = rsqrtf(var + LN_EPS);
    const float invq = 1.0f / qscale[0];

    float* orow = out + (long long)(m0 + g) * DOUT;
    for (int j = lt * 4; j < DOUT; j += 64) {
        uint2 pk = *(const uint2*)(row + j);
        float4 gm = *(const float4*)(gamma + j);
        float4 bt = *(const float4*)(beta + j);
        float v0 = __uint_as_float(pk.x << 16);
        float v1 = __uint_as_float(pk.x & 0xFFFF0000u);
        float v2 = __uint_as_float(pk.y << 16);
        float v3 = __uint_as_float(pk.y & 0xFFFF0000u);
        float4 o;
        o.x = rintf(tanhf(((v0 - mean) * rstd * gm.x + bt.x) * invq) * 127.0f) * (1.0f / 127.0f);
        o.y = rintf(tanhf(((v1 - mean) * rstd * gm.y + bt.y) * invq) * 127.0f) * (1.0f / 127.0f);
        o.z = rintf(tanhf(((v2 - mean) * rstd * gm.z + bt.z) * invq) * 127.0f) * (1.0f / 127.0f);
        o.w = rintf(tanhf(((v3 - mean) * rstd * gm.w + bt.w) * invq) * 127.0f) * (1.0f / 127.0f);
        *(float4*)(orow + j) = o;
    }
}

// ---------------------------------------------------------------------------
extern "C" void kernel_launch(void* const* d_in, const int* in_sizes, int n_in,
                              void* d_out, int out_size, void* d_ws, size_t ws_size,
                              hipStream_t stream) {
    const float* x     = (const float*)d_in[0];
    const float* w     = (const float*)d_in[1];
    const float* bias  = (const float*)d_in[2];
    const float* scale = (const float*)d_in[3];
    const float* gamma = (const float*)d_in[4];
    const float* beta  = (const float*)d_in[5];
    const float* qs    = (const float*)d_in[6];
    float* out = (float*)d_out;

    const long long NX = (long long)BDIM * SDIM * DIN;   // 67,108,864
    const long long NW = (long long)DOUT * DIN;          // 16,777,216

    unsigned short* xb = (unsigned short*)d_ws;                          // 128 MB
    unsigned short* wb = (unsigned short*)((char*)d_ws + NX * 2);        // +32 MB

    // 1) x -> bf16
    {
        int n8 = (int)(NX / 8);
        cvt_x_bf16_kernel<<<(n8 + 255) / 256, 256, 0, stream>>>(x, xb, n8);
    }
    // 2) W -> ternary bf16
    {
        int n8 = (int)(NW / 8);
        ternarize_w_kernel<<<(n8 + 255) / 256, 256, 0, stream>>>(w, wb, n8);
    }
    // 3) fused GEMM + LN + quantize
    {
        const int blocks = (BDIM * SDIM) / MROWS;            // 512
        const size_t lds = (size_t)MROWS * DOUT * sizeof(unsigned short); // 256 KB
        bitnet_gemm_ln_kernel<<<blocks, TPB, lds, stream>>>(
            xb, wb, bias, scale, gamma, beta, qs, out);
    }
}
